// TransformerEncoderLayer_71305047048344
// MI455X (gfx1250) — matrix operand order repaired and checked by reading, not compiled, not run on hardware
//
#include <hip/hip_runtime.h>

// ---------------------------------------------------------------------------
// Types for CDNA5 WMMA (wave32): D(16x16,f32) = A(16x32,bf16) x B(32x16,bf16) + C
// ---------------------------------------------------------------------------
typedef __attribute__((ext_vector_type(16))) __bf16 v16bf;
typedef __attribute__((ext_vector_type(8)))  __bf16 v8bf;
typedef __attribute__((ext_vector_type(8)))  float  v8f;

#define WMMA_BF16(a, b, c) \
  __builtin_amdgcn_wmma_f32_16x16x32_bf16(false, (a), false, (b), (short)0, (c), false, false)

// Concatenate two 8-element bf16 chunks (16B-aligned b128 loads) into the
// 16-element per-lane fragment register block. Works for LDS or global ptrs.
__device__ __forceinline__ v16bf frag_cat(const __bf16* p0, const __bf16* p1) {
  v8bf lo = *(const v8bf*)p0;
  v8bf hi = *(const v8bf*)p1;
  v16bf r;
#pragma unroll
  for (int i = 0; i < 8; ++i) { r[i] = lo[i]; r[i + 8] = hi[i]; }
  return r;
}

union BF4 { __bf16 h[4]; uint2 u; };

__device__ __forceinline__ void store_bf16x4v(__bf16* d, float a, float b, float c, float e) {
  BF4 p;
  p.h[0] = (__bf16)a; p.h[1] = (__bf16)b; p.h[2] = (__bf16)c; p.h[3] = (__bf16)e;
  *(uint2*)d = p.u;
}

// xor-butterfly within 16-lane rows via v_permlane16_b32 (VALU, no DS waits).
__device__ __forceinline__ float xorp16(float v, unsigned s0, unsigned s1) {
  unsigned u = __builtin_bit_cast(unsigned, v);
  u = __builtin_amdgcn_permlane16(u, u, s0, s1, false, false);
  return __builtin_bit_cast(float, u);
}
#define XSEL1_LO 0x67452301u
#define XSEL1_HI 0xEFCDAB89u
#define XSEL2_LO 0x54761032u
#define XSEL2_HI 0xDCFE98BAu
#define XSEL4_LO 0x32107654u
#define XSEL4_HI 0xBA98FEDCu
#define XSEL8_LO 0xFEDCBA98u
#define XSEL8_HI 0x76543210u

__device__ __forceinline__ float rowmax16(float v) {
  v = fmaxf(v, xorp16(v, XSEL1_LO, XSEL1_HI));
  v = fmaxf(v, xorp16(v, XSEL2_LO, XSEL2_HI));
  v = fmaxf(v, xorp16(v, XSEL4_LO, XSEL4_HI));
  v = fmaxf(v, xorp16(v, XSEL8_LO, XSEL8_HI));
  return v;
}
__device__ __forceinline__ float rowsum16(float v) {
  v += xorp16(v, XSEL1_LO, XSEL1_HI);
  v += xorp16(v, XSEL2_LO, XSEL2_HI);
  v += xorp16(v, XSEL4_LO, XSEL4_HI);
  v += xorp16(v, XSEL8_LO, XSEL8_HI);
  return v;
}

// ---------------------------------------------------------------------------
// Prep kernels (run once per launch; pure bandwidth, ~5us total at 23TB/s).
// ---------------------------------------------------------------------------
// Elementwise f32 -> bf16 (4 elements / thread).
__global__ __launch_bounds__(256) void cvt_bf16_kernel(
    const float* __restrict__ in, __bf16* __restrict__ out) {
  const size_t i = (size_t)blockIdx.x * 256 + threadIdx.x;
  float4 f = ((const float4*)in)[i];
  store_bf16x4v(out + i * 4, f.x, f.y, f.z, f.w);
}

// Tiled transpose + convert: src f32 [K,N] row-major -> dst bf16 [N,K].
__global__ __launch_bounds__(256) void tconv_kernel(
    const float* __restrict__ src, __bf16* __restrict__ dst, int K, int N) {
  __shared__ float tile[32][33];
  const int tx = threadIdx.x & 31;
  const int ty = threadIdx.x >> 5;  // 0..7
  const int k0 = blockIdx.y * 32;
  const int n0 = blockIdx.x * 32;
#pragma unroll
  for (int i = 0; i < 32; i += 8)
    tile[ty + i][tx] = src[(size_t)(k0 + ty + i) * N + n0 + tx];
  __syncthreads();
#pragma unroll
  for (int i = 0; i < 32; i += 8)
    dst[(size_t)(n0 + ty + i) * K + k0 + tx] = (__bf16)tile[tx][ty + i];
}

// ---------------------------------------------------------------------------
// GEMM: out[M,N] = A[M,K](bf16) @ Bt[N,K](bf16)^T + bias (+res f32) (+gelu).
// Block = 256 threads = 8 waves, tile 128x128, KT = 32, double-buffered LDS
// with register prefetch (one barrier per K-step). Staging is pure b128
// copies: no conversions, no transposes in the hot loop.
// ---------------------------------------------------------------------------
template <bool RES, bool GELU, bool OUTBF>
__global__ __launch_bounds__(256) void gemm_bf16_kernel(
    const __bf16* __restrict__ A, const __bf16* __restrict__ Bt,
    const float* __restrict__ bias, const float* __restrict__ res,
    void* __restrict__ outp, int M, int N, int K) {
  // Row stride 40 bf16 = 80 bytes (multiple of 16 -> aligned ds b128 ops).
  __shared__ __align__(16) __bf16 As[2][128][40];  // [m][k]
  __shared__ __align__(16) __bf16 Bs[2][128][40];  // [n][k]

  const int t    = threadIdx.x;
  const int lane = t & 31;
  const int w    = t >> 5;
  const int h0   = lane >> 4;
  const int l15  = lane & 15;
  const int wm   = w >> 1;      // 0..3
  const int wn   = w & 1;       // 0..1
  const int bm   = blockIdx.y * 128;
  const int bn   = blockIdx.x * 128;

  v8f acc[2][4] = {};

  // Staging: thread -> (tile row, 16-element chunk); 32B per thread per tile.
  const int srow = t >> 1;        // 0..127
  const int skp  = (t & 1) * 16;  // 0 / 16

  uint4 a_reg[2], b_reg[2];
  auto load_global = [&](int k0) {
    const __bf16* ga = A  + (size_t)(bm + srow) * K + (k0 + skp);
    const __bf16* gb = Bt + (size_t)(bn + srow) * K + (k0 + skp);
    a_reg[0] = *(const uint4*)ga;  a_reg[1] = *(const uint4*)(ga + 8);
    b_reg[0] = *(const uint4*)gb;  b_reg[1] = *(const uint4*)(gb + 8);
  };
  auto store_lds = [&](int buf) {
    *(uint4*)&As[buf][srow][skp]     = a_reg[0];
    *(uint4*)&As[buf][srow][skp + 8] = a_reg[1];
    *(uint4*)&Bs[buf][srow][skp]     = b_reg[0];
    *(uint4*)&Bs[buf][srow][skp + 8] = b_reg[1];
  };

  load_global(0);
  store_lds(0);

  const int nk = K >> 5;
  for (int ki = 0; ki < nk; ++ki) {
    __syncthreads();
    const int buf  = ki & 1;
    const bool more = (ki + 1) < nk;
    if (more) load_global((ki + 1) << 5);  // overlap with WMMAs below

    v16bf af[2], bf[4];
#pragma unroll
    for (int mt = 0; mt < 2; ++mt) {
      // A lane layout (ISA 7.12.2): K = {h0*8..+7, 16+h0*8..+7}
      const __bf16* p = &As[buf][wm * 32 + mt * 16 + l15][h0 * 8];
      af[mt] = frag_cat(p, p + 16);
    }
#pragma unroll
    for (int nt = 0; nt < 4; ++nt) {
      // B lane layout: K = {h0*8..+7, 16+h0*8..+7} (same as A)
      const __bf16* p = &Bs[buf][wn * 64 + nt * 16 + l15][h0 * 8];
      bf[nt] = frag_cat(p, p + 16);
    }
#pragma unroll
    for (int mt = 0; mt < 2; ++mt)
#pragma unroll
      for (int nt = 0; nt < 4; ++nt)
        acc[mt][nt] = WMMA_BF16(af[mt], bf[nt], acc[mt][nt]);

    if (more) store_lds(buf ^ 1);  // buf^1 last read before this barrier
  }

  // Epilogue: C layout -> VGPR r holds (m = r + 8*h0, n = l15) per n-tile.
#pragma unroll
  for (int mt = 0; mt < 2; ++mt) {
#pragma unroll
    for (int nt = 0; nt < 4; ++nt) {
      const int gn   = bn + wn * 64 + nt * 16 + l15;
      const float bv = bias[gn];
#pragma unroll
      for (int r = 0; r < 8; ++r) {
        const int gm = bm + wm * 32 + mt * 16 + h0 * 8 + r;
        float v = acc[mt][nt][r] + bv;
        if (RES) v += res[(size_t)gm * N + gn];
        if (GELU) v = 0.5f * v * (1.0f + erff(v * 0.7071067811865475f));
        if (OUTBF) ((__bf16*)outp)[(size_t)gm * N + gn] = (__bf16)v;
        else       ((float*)outp)[(size_t)gm * N + gn] = v;
      }
    }
  }
}

// ---------------------------------------------------------------------------
// Flash attention on bf16 QKV. qkv layout: [B*S, 3072] bf16; per head h:
// q at h*192, k at h*192+64, v at h*192+128. Grid (S/128, H, B); 8 waves,
// each owning 16 query rows. Q fragments load directly from global (no LDS);
// K/V double-buffered in LDS (one barrier/iter) with register prefetch.
// 1/sqrt(hd) folds into the softmax. Output bf16 [B*S, 1024].
// ---------------------------------------------------------------------------
__global__ __launch_bounds__(256) void attn_kernel(
    const __bf16* __restrict__ qkv, __bf16* __restrict__ out) {
  __shared__ __align__(16) __bf16 Ks[2][64][72];    // [key][d]  (B for Q.K^T)
  __shared__ __align__(16) __bf16 Vt[2][64][72];    // [d][key]  (B for P.V)
  __shared__ __align__(16) __bf16 Ps[8][16][72];    // per-wave P relayout

  const int t    = threadIdx.x;
  const int lane = t & 31;
  const int w    = t >> 5;
  const int h0   = lane >> 4;
  const int l15  = lane & 15;
  const int b    = blockIdx.z;
  const int h    = blockIdx.y;
  const int q0   = blockIdx.x * 128;

  const size_t rs   = 3072;
  const size_t hoff = (size_t)h * 192;

  // Q fragments straight from global (bf16, A-layout addressing).
  v16bf qf[2];
  {
    const __bf16* qp = qkv + (size_t)(b * 2048 + q0 + w * 16 + l15) * rs + hoff;
#pragma unroll
    for (int kk = 0; kk < 2; ++kk)
      qf[kk] = frag_cat(qp + kk * 32 + h0 * 8, qp + kk * 32 + h0 * 8 + 16);
  }

  v8f o[4] = {};
  float mrow[8], lrow[8];
#pragma unroll
  for (int r = 0; r < 8; ++r) { mrow[r] = -1e30f; lrow[r] = 0.f; }

  // K staging: thread -> (key row, 16-d chunk): 2 x b128 copy.
  const int jr = t >> 2;         // 0..63
  const int dp = (t & 3) * 16;   // 0..48
  // V staging: thread gathers 4 keys x 4 d and transposes in registers.
  const int vdc = (t & 15) * 4;  // d   0..60
  const int vjg = (t >> 4) * 4;  // key 0..60

  uint4 kr[2];
  BF4   vr[4];
  auto load_kv = [&](int j0) {
    const __bf16* gk = qkv + (size_t)(b * 2048 + j0 + jr) * rs + hoff + 64 + dp;
    kr[0] = *(const uint4*)gk;
    kr[1] = *(const uint4*)(gk + 8);
    const __bf16* gv = qkv + (size_t)(b * 2048 + j0 + vjg) * rs + hoff + 128 + vdc;
#pragma unroll
    for (int i = 0; i < 4; ++i) vr[i].u = *(const uint2*)(gv + (size_t)i * rs);
  };
  auto store_kv = [&](int buf) {
    *(uint4*)&Ks[buf][jr][dp]     = kr[0];
    *(uint4*)&Ks[buf][jr][dp + 8] = kr[1];
#pragma unroll
    for (int c = 0; c < 4; ++c) {
      BF4 oo;
      oo.h[0] = vr[0].h[c]; oo.h[1] = vr[1].h[c];
      oo.h[2] = vr[2].h[c]; oo.h[3] = vr[3].h[c];
      *(uint2*)&Vt[buf][vdc + c][vjg] = oo.u;
    }
  };

  load_kv(0);
  store_kv(0);

  for (int jb = 0; jb < 32; ++jb) {
    __syncthreads();
    const int buf  = jb & 1;
    const bool more = (jb + 1) < 32;
    if (more) load_kv((jb + 1) * 64);  // overlap with compute below

    // S_raw = Q @ K^T : 16 queries x 64 keys (hd contraction, 2 WMMA steps).
    v8f c[4] = {};
#pragma unroll
    for (int nt = 0; nt < 4; ++nt) {
#pragma unroll
      for (int kk = 0; kk < 2; ++kk) {
        const __bf16* p = &Ks[buf][nt * 16 + l15][kk * 32 + h0 * 8];
        v16bf kb = frag_cat(p, p + 16);
        c[nt] = WMMA_BF16(qf[kk], kb, c[nt]);
      }
    }

    // Online softmax with scale 0.125 folded in; row r+8*h0 spans the 16
    // lanes of this half-wave.
#pragma unroll
    for (int r = 0; r < 8; ++r) {
      float mx = fmaxf(fmaxf(c[0][r], c[1][r]), fmaxf(c[2][r], c[3][r]));
      mx = rowmax16(mx) * 0.125f;
      const float mnew  = fmaxf(mrow[r], mx);
      const float alpha = __expf(mrow[r] - mnew);
      mrow[r] = mnew;
      float ps = 0.f;
#pragma unroll
      for (int nt = 0; nt < 4; ++nt) {
        const float p = __expf(fmaf(c[nt][r], 0.125f, -mnew));
        c[nt][r] = p;
        ps += p;
      }
      ps = rowsum16(ps);
      lrow[r] = lrow[r] * alpha + ps;
#pragma unroll
      for (int dt = 0; dt < 4; ++dt) o[dt][r] *= alpha;
    }

    // Relayout P (C-fragment -> A-fragment) through per-wave LDS buffer.
#pragma unroll
    for (int nt = 0; nt < 4; ++nt)
#pragma unroll
      for (int r = 0; r < 8; ++r)
        Ps[w][h0 * 8 + r][nt * 16 + l15] = (__bf16)c[nt][r];

    // O += P @ V (contraction over 64 keys = 2 WMMA steps).
#pragma unroll
    for (int kk = 0; kk < 2; ++kk) {
      const __bf16* pp = &Ps[w][l15][kk * 32 + h0 * 8];
      v16bf pf = frag_cat(pp, pp + 16);
#pragma unroll
      for (int dt = 0; dt < 4; ++dt) {
        const __bf16* vp = &Vt[buf][dt * 16 + l15][kk * 32 + h0 * 8];
        v16bf vb = frag_cat(vp, vp + 16);
        o[dt] = WMMA_BF16(pf, vb, o[dt]);
      }
    }

    if (more) store_kv(buf ^ 1);  // buf^1 last read before this barrier
  }

  // O / l -> bf16 attn output [B*S, 1024], head-major columns.
#pragma unroll
  for (int dt = 0; dt < 4; ++dt) {
    const int col = h * 64 + dt * 16 + l15;
#pragma unroll
    for (int r = 0; r < 8; ++r) {
      const int qrow = q0 + w * 16 + h0 * 8 + r;
      out[(size_t)(b * 2048 + qrow) * 1024 + col] = (__bf16)(o[dt][r] / lrow[r]);
    }
  }
}

// ---------------------------------------------------------------------------
// LayerNorm over D=1024; one 256-thread block per row. DUAL also emits a
// bf16 copy (GEMM A operand) alongside the f32 result (residual path).
// ---------------------------------------------------------------------------
template <bool DUAL>
__global__ __launch_bounds__(256) void ln_kernel(
    const float* __restrict__ in, const float* __restrict__ g,
    const float* __restrict__ be, float* __restrict__ out,
    __bf16* __restrict__ out2) {
  __shared__ float sm[8];
  __shared__ float sv[8];
  const int t    = threadIdx.x;
  const int lane = t & 31;
  const int w    = t >> 5;
  const size_t row = blockIdx.x;

  float4 v = ((const float4*)(in + row * 1024))[t];
  float s = v.x + v.y + v.z + v.w;
  s += __shfl_xor(s, 1, 32);  s += __shfl_xor(s, 2, 32);
  s += __shfl_xor(s, 4, 32);  s += __shfl_xor(s, 8, 32);
  s += __shfl_xor(s, 16, 32);
  if (lane == 0) sm[w] = s;
  __syncthreads();
  float mean = 0.f;
#pragma unroll
  for (int i = 0; i < 8; ++i) mean += sm[i];
  mean *= (1.f / 1024.f);

  const float dx = v.x - mean, dy = v.y - mean, dz = v.z - mean, dw = v.w - mean;
  float q = dx * dx + dy * dy + dz * dz + dw * dw;
  q += __shfl_xor(q, 1, 32);  q += __shfl_xor(q, 2, 32);
  q += __shfl_xor(q, 4, 32);  q += __shfl_xor(q, 8, 32);
  q += __shfl_xor(q, 16, 32);
  if (lane == 0) sv[w] = q;
  __syncthreads();
  float var = 0.f;
#pragma unroll
  for (int i = 0; i < 8; ++i) var += sv[i];
  var *= (1.f / 1024.f);
  const float inv = rsqrtf(var + 1e-5f);

  float4 gg = ((const float4*)g)[t];
  float4 bb = ((const float4*)be)[t];
  float4 r;
  r.x = dx * inv * gg.x + bb.x;
  r.y = dy * inv * gg.y + bb.y;
  r.z = dz * inv * gg.z + bb.z;
  r.w = dw * inv * gg.w + bb.w;
  ((float4*)(out + row * 1024))[t] = r;
  if (DUAL) store_bf16x4v(out2 + row * 1024 + t * 4, r.x, r.y, r.z, r.w);
}

// ---------------------------------------------------------------------------
// Launch: prep(cvt x, transpose weights) -> qkv -> attn -> proj+res -> LN1
//         -> FFN1(gelu) -> FFN2+res -> LN2
// ---------------------------------------------------------------------------
extern "C" void kernel_launch(void* const* d_in, const int* in_sizes, int n_in,
                              void* d_out, int out_size, void* d_ws, size_t ws_size,
                              hipStream_t stream) {
  (void)in_sizes; (void)n_in; (void)out_size; (void)ws_size;
  const float* x    = (const float*)d_in[0];
  const float* Wqkv = (const float*)d_in[1];
  const float* bqkv = (const float*)d_in[2];
  const float* Wo   = (const float*)d_in[3];
  const float* bo   = (const float*)d_in[4];
  const float* W1   = (const float*)d_in[5];
  const float* b1   = (const float*)d_in[6];
  const float* W2   = (const float*)d_in[7];
  const float* b2   = (const float*)d_in[8];
  const float* g1   = (const float*)d_in[9];
  const float* be1  = (const float*)d_in[10];
  const float* g2   = (const float*)d_in[11];
  const float* be2  = (const float*)d_in[12];
  float* out = (float*)d_out;

  char* ws = (char*)d_ws;
  const size_t MB = 1024 * 1024;
  __bf16* qkv   = (__bf16*)(ws + 0);        // [4096,3072] bf16  24MB
  __bf16* att   = (__bf16*)(ws + 24 * MB);  // [4096,1024] bf16   8MB
  float*  y1    = (float*)(ws + 32 * MB);   // [4096,1024] f32   16MB
  float*  x1    = (float*)(ws + 48 * MB);   // [4096,1024] f32   16MB
  __bf16* x1b   = (__bf16*)(ws + 64 * MB);  // [4096,1024] bf16   8MB
  __bf16* Wqkvt = (__bf16*)(ws + 72 * MB);  // [3072,1024] bf16   6MB
  __bf16* Wot   = (__bf16*)(ws + 78 * MB);  // [1024,1024] bf16   2MB
  __bf16* W1t   = (__bf16*)(ws + 80 * MB);  // [4096,1024] bf16   8MB
  __bf16* W2t   = (__bf16*)(ws + 88 * MB);  // [1024,4096] bf16   8MB
  __bf16* xb    = (__bf16*)(ws + 96 * MB);  // [4096,1024] bf16   8MB
  __bf16* hbuf  = (__bf16*)(ws + 0);        // [4096,4096] bf16  32MB (reuse qkv+att)
  float*  y2    = (float*)(ws + 32 * MB);   // [4096,1024] f32   16MB (reuse y1)

  const dim3 blk(256);

  // 0) Prep: x -> bf16; weights -> transposed bf16 [N][K].
  cvt_bf16_kernel<<<dim3(4096), blk, 0, stream>>>(x, xb);
  tconv_kernel<<<dim3(96, 32), blk, 0, stream>>>(Wqkv, Wqkvt, 1024, 3072);
  tconv_kernel<<<dim3(32, 32), blk, 0, stream>>>(Wo, Wot, 1024, 1024);
  tconv_kernel<<<dim3(128, 32), blk, 0, stream>>>(W1, W1t, 1024, 4096);
  tconv_kernel<<<dim3(32, 128), blk, 0, stream>>>(W2, W2t, 4096, 1024);

  // 1) QKV projection -> bf16
  gemm_bf16_kernel<false, false, true><<<dim3(24, 32), blk, 0, stream>>>(
      xb, Wqkvt, bqkv, nullptr, qkv, 4096, 3072, 1024);

  // 2) Multi-head flash attention -> bf16
  attn_kernel<<<dim3(16, 16, 2), blk, 0, stream>>>(qkv, att);

  // 3) Output projection + residual(x) -> f32
  gemm_bf16_kernel<true, false, false><<<dim3(8, 32), blk, 0, stream>>>(
      att, Wot, bo, x, y1, 4096, 1024, 1024);

  // 4) LayerNorm 1 -> f32 (residual) + bf16 (GEMM operand)
  ln_kernel<true><<<dim3(4096), blk, 0, stream>>>(y1, g1, be1, x1, x1b);

  // 5) FFN up + exact GELU -> bf16
  gemm_bf16_kernel<false, true, true><<<dim3(32, 32), blk, 0, stream>>>(
      x1b, W1t, b1, nullptr, hbuf, 4096, 4096, 1024);

  // 6) FFN down + residual(x1) -> f32
  gemm_bf16_kernel<true, false, false><<<dim3(8, 32), blk, 0, stream>>>(
      hbuf, W2t, b2, x1, y2, 4096, 1024, 4096);

  // 7) LayerNorm 2 -> output f32
  ln_kernel<false><<<dim3(4096), blk, 0, stream>>>(y2, g2, be2, out, nullptr);
}
